// GATskip_90512140796493
// MI455X (gfx1250) — matrix operand order repaired and checked by reading, not compiled, not run on hardware
//
#include <hip/hip_runtime.h>

#define DH 128            // feature dim
#define NCOLS 512         // 4 fused output matrices (K|Q|V|S)
#define LDS_STRIDE 136    // 128 + 8 pad (bf16 units) -> conflict-free b128 LDS reads

typedef __attribute__((ext_vector_type(16))) __bf16 v16bf;
typedef __attribute__((ext_vector_type(8)))  __bf16 v8bf;
typedef __attribute__((ext_vector_type(8)))  float  v8f;

__device__ __forceinline__ unsigned short f2bf(float f) {
    unsigned int u = __float_as_uint(f);
    u += 0x7FFFu + ((u >> 16) & 1u);   // round-to-nearest-even
    return (unsigned short)(u >> 16);
}

__device__ __forceinline__ float sigmoidf(float z) {
    return 1.0f / (1.0f + __expf(-z));
}

// ---------------------------------------------------------------------------
// Pack [Wk|Wq|Wv|Ws] (each fp32 [128,128], row-major [k][n]) into bf16 WMMA
// B-fragment layout. Index t = ((nt*4 + kt)*32 + lane)*16 + i
//   lanes 0-15 : K = kt*32 + i      , N = nt*16 + lane
//   lanes 16-31: K = kt*32 + 16 + i , N = nt*16 + lane-16
// ---------------------------------------------------------------------------
__global__ __launch_bounds__(256) void pack_weights(
    const float* __restrict__ Wk, const float* __restrict__ Wq,
    const float* __restrict__ Wv, const float* __restrict__ Ws,
    unsigned short* __restrict__ Wp)
{
    int t = blockIdx.x * 256 + threadIdx.x;
    if (t >= 32 * 4 * 512) return;
    int i    = t & 15;
    int lane = (t >> 4) & 31;
    int kt   = (t >> 9) & 3;
    int nt   = t >> 11;                       // 0..31
    int K    = kt * 32 + i + ((lane >> 4) << 4);
    int ncg  = nt * 16 + (lane & 15);         // 0..511
    int mat  = ncg >> 7;
    int col  = ncg & 127;
    const float* W = (mat == 0) ? Wk : (mat == 1) ? Wq : (mat == 2) ? Wv : Ws;
    Wp[t] = f2bf(W[K * DH + col]);
}

// ---------------------------------------------------------------------------
// Fused GEMM: [K|Q|V|S] = X @ Wp (+bias on S columns). One 16-row tile/block,
// 8 waves x 4 column-tiles, K-loop 4 x v_wmma_f32_16x16x32_bf16.
// ---------------------------------------------------------------------------
__global__ __launch_bounds__(256) void gemm_kqvs(
    const float* __restrict__ X, const unsigned short* __restrict__ Wp,
    const float* __restrict__ bias,
    float* __restrict__ Kb, float* __restrict__ Qb,
    float* __restrict__ Vb, float* __restrict__ Sb, int nNodes)
{
    __shared__ __align__(16) unsigned short sA[16 * LDS_STRIDE];
    const int row0 = blockIdx.x * 16;
    const int tid  = threadIdx.x;

    // Stage 16x128 X tile into LDS as bf16
    for (int r = tid; r < 16 * DH; r += 256) {
        int m = r >> 7, k = r & 127;
        int row = row0 + m;
        float v = (row < nNodes) ? X[(size_t)row * DH + k] : 0.0f;
        sA[m * LDS_STRIDE + k] = f2bf(v);
    }
    __syncthreads();

    const int wave = tid >> 5;
    const int lane = tid & 31;
    const int half = lane >> 4;     // 0: lanes 0-15, 1: lanes 16-31
    const int lm   = lane & 15;

    for (int nt4 = 0; nt4 < 4; ++nt4) {
        const int nt = wave * 4 + nt4;                 // 0..31 column tile
        v8f acc = {};
        const unsigned short* wp = Wp + ((size_t)nt * 4) * 512;
        #pragma unroll
        for (int kt = 0; kt < 4; ++kt) {
            // A fragment, contiguous k order (k = half*16 + s, same as B): two 16B chunks
            const unsigned short* ap = &sA[lm * LDS_STRIDE + kt * 32 + half * 16];
            union { v16bf v; v8bf h[2]; } au;
            au.h[0] = *(const v8bf*)(ap);
            au.h[1] = *(const v8bf*)(ap + 8);
            // B fragment: pre-swizzled, contiguous 32B per lane
            v16bf bfrag = *(const v16bf*)(wp + kt * 512 + lane * 16);
            acc = __builtin_amdgcn_wmma_f32_16x16x32_bf16(
                false, au.v, false, bfrag, (short)0, acc, false, false);
        }
        // C layout: VGPR r -> row (r + half*8), col = lane%16
        const int ncg = nt * 16 + lm;
        const int mat = ncg >> 7;
        const int col = ncg & 127;
        float* dst = (mat == 0) ? Kb : (mat == 1) ? Qb : (mat == 2) ? Vb : Sb;
        const float badd = (mat == 3) ? bias[col] : 0.0f;
        #pragma unroll
        for (int r = 0; r < 8; ++r) {
            int row = row0 + r + half * 8;
            if (row < nNodes) dst[(size_t)row * DH + col] = acc[r] + badd;
        }
    }
}

// ---------------------------------------------------------------------------
// Edge pass: one wave per edge, float4 per lane (32 x 16B = full 128-f row).
// agg[dst] += sigmoid(k[dst]+q[src]) * v[src]
// ---------------------------------------------------------------------------
__global__ __launch_bounds__(256) void edge_gate(
    const int* __restrict__ src, const int* __restrict__ dst,
    const float* __restrict__ Kb, const float* __restrict__ Qb,
    const float* __restrict__ Vb, float* __restrict__ AGG, int nEdges)
{
    long long t = (long long)blockIdx.x * 256 + threadIdx.x;
    int e    = (int)(t >> 5);
    int lane = (int)(t & 31);
    if (e >= nEdges) return;
    int s = src[e], d = dst[e];
    const float4 k = *(const float4*)&Kb[(size_t)d * DH + lane * 4];
    const float4 q = *(const float4*)&Qb[(size_t)s * DH + lane * 4];
    const float4 v = *(const float4*)&Vb[(size_t)s * DH + lane * 4];
    float* o = &AGG[(size_t)d * DH + lane * 4];
    atomicAdd(o + 0, sigmoidf(k.x + q.x) * v.x);
    atomicAdd(o + 1, sigmoidf(k.y + q.y) * v.y);
    atomicAdd(o + 2, sigmoidf(k.z + q.z) * v.z);
    atomicAdd(o + 3, sigmoidf(k.w + q.w) * v.w);
}

__global__ __launch_bounds__(256) void relu4(
    const float* __restrict__ in, float* __restrict__ out, long long n4)
{
    long long t = (long long)blockIdx.x * 256 + threadIdx.x;
    if (t >= n4) return;
    float4 v = *(const float4*)&in[t * 4];
    v.x = fmaxf(v.x, 0.0f); v.y = fmaxf(v.y, 0.0f);
    v.z = fmaxf(v.z, 0.0f); v.w = fmaxf(v.w, 0.0f);
    *(float4*)&out[t * 4] = v;
}

// ---------------------------------------------------------------------------
// Layer 3 (Dout=1): wave per node, 4 scalar dots + wave reduction.
// out[node] initialized to Ws3.x + b3 here; edge3 accumulates on top.
// ---------------------------------------------------------------------------
__global__ __launch_bounds__(256) void node_dots(
    const float* __restrict__ X,
    const float* __restrict__ Wk, const float* __restrict__ Wq,
    const float* __restrict__ Wv, const float* __restrict__ Ws,
    const float* __restrict__ b,
    float* __restrict__ k3, float* __restrict__ q3, float* __restrict__ v3,
    float* __restrict__ out, int nNodes)
{
    long long t = (long long)blockIdx.x * 256 + threadIdx.x;
    int node = (int)(t >> 5);
    int lane = (int)(t & 31);
    if (node >= nNodes) return;
    const float4 x  = *(const float4*)&X[(size_t)node * DH + lane * 4];
    const float4 wk = *(const float4*)&Wk[lane * 4];
    const float4 wq = *(const float4*)&Wq[lane * 4];
    const float4 wv = *(const float4*)&Wv[lane * 4];
    const float4 ws = *(const float4*)&Ws[lane * 4];
    float ak = x.x*wk.x + x.y*wk.y + x.z*wk.z + x.w*wk.w;
    float aq = x.x*wq.x + x.y*wq.y + x.z*wq.z + x.w*wq.w;
    float av = x.x*wv.x + x.y*wv.y + x.z*wv.z + x.w*wv.w;
    float as = x.x*ws.x + x.y*ws.y + x.z*ws.z + x.w*ws.w;
    #pragma unroll
    for (int off = 16; off > 0; off >>= 1) {
        ak += __shfl_down(ak, off, 32);
        aq += __shfl_down(aq, off, 32);
        av += __shfl_down(av, off, 32);
        as += __shfl_down(as, off, 32);
    }
    if (lane == 0) {
        k3[node] = ak; q3[node] = aq; v3[node] = av;
        out[node] = as + b[0];
    }
}

__global__ __launch_bounds__(256) void edge_scalar(
    const int* __restrict__ src, const int* __restrict__ dst,
    const float* __restrict__ k3, const float* __restrict__ q3,
    const float* __restrict__ v3, float* __restrict__ out, int nEdges)
{
    int e = blockIdx.x * 256 + threadIdx.x;
    if (e >= nEdges) return;
    int s = src[e], d = dst[e];
    atomicAdd(&out[d], sigmoidf(k3[d] + q3[s]) * v3[s]);
}

// ---------------------------------------------------------------------------
extern "C" void kernel_launch(void* const* d_in, const int* in_sizes, int n_in,
                              void* d_out, int out_size, void* d_ws, size_t ws_size,
                              hipStream_t stream) {
    const float* x   = (const float*)d_in[0];
    const int*   ei  = (const int*)d_in[1];
    const int N = in_sizes[0] / DH;
    const int E = in_sizes[1] / 2;
    const int* src = ei;
    const int* dst = ei + E;

    const float* Wk1 = (const float*)d_in[2];
    const float* Wq1 = (const float*)d_in[3];
    const float* Wv1 = (const float*)d_in[4];
    const float* Ws1 = (const float*)d_in[5];
    const float* b1  = (const float*)d_in[6];
    const float* Wk2 = (const float*)d_in[7];
    const float* Wq2 = (const float*)d_in[8];
    const float* Wv2 = (const float*)d_in[9];
    const float* Ws2 = (const float*)d_in[10];
    const float* b2  = (const float*)d_in[11];
    const float* Wk3 = (const float*)d_in[12];
    const float* Wq3 = (const float*)d_in[13];
    const float* Wv3 = (const float*)d_in[14];
    const float* Ws3 = (const float*)d_in[15];
    const float* b3  = (const float*)d_in[16];

    float* ws = (float*)d_ws;
    const size_t NB = (size_t)N * DH;         // floats per feature buffer
    float* Kb  = ws;
    float* Qb  = ws + NB;
    float* Vb  = ws + 2 * NB;
    float* AGG = ws + 3 * NB;
    float* XN  = ws + 4 * NB;
    float* k3  = ws + 5 * NB;
    float* q3  = k3 + N;
    float* v3  = q3 + N;
    unsigned short* Wp1 = (unsigned short*)(v3 + N);
    unsigned short* Wp2 = Wp1 + 32 * 4 * 512;

    const int gblocks = (N + 15) / 16;
    const int eblocks = (int)(((long long)E * 32 + 255) / 256);
    const long long n4 = (long long)NB / 4;
    const int rblocks = (int)((n4 + 255) / 256);
    const int dblocks = (int)(((long long)N * 32 + 255) / 256);
    const int sblocks = (E + 255) / 256;

    pack_weights<<<256, 256, 0, stream>>>(Wk1, Wq1, Wv1, Ws1, Wp1);
    pack_weights<<<256, 256, 0, stream>>>(Wk2, Wq2, Wv2, Ws2, Wp2);

    // Layer 1
    gemm_kqvs<<<gblocks, 256, 0, stream>>>(x, Wp1, b1, Kb, Qb, Vb, AGG, N);
    edge_gate<<<eblocks, 256, 0, stream>>>(src, dst, Kb, Qb, Vb, AGG, E);
    relu4<<<rblocks, 256, 0, stream>>>(AGG, XN, n4);

    // Layer 2
    gemm_kqvs<<<gblocks, 256, 0, stream>>>(XN, Wp2, b2, Kb, Qb, Vb, AGG, N);
    edge_gate<<<eblocks, 256, 0, stream>>>(src, dst, Kb, Qb, Vb, AGG, E);
    relu4<<<rblocks, 256, 0, stream>>>(AGG, XN, n4);

    // Layer 3 (Dout = 1)
    node_dots<<<dblocks, 256, 0, stream>>>(XN, Wk3, Wq3, Wv3, Ws3, b3,
                                           k3, q3, v3, (float*)d_out, N);
    edge_scalar<<<sblocks, 256, 0, stream>>>(src, dst, k3, q3, v3,
                                             (float*)d_out, E);
}
